// MultiHeadAttention_47888885351101
// MI455X (gfx1250) — matrix operand order repaired and checked by reading, not compiled, not run on hardware
//
#include <hip/hip_runtime.h>
#include <hip/hip_bf16.h>
#include <math.h>

// ---------------------------------------------------------------------------
// MI455X / gfx1250 multi-head attention, f16 WMMA (16x16x32, f32 accum)
// + Tensor Data Mover staging for the f16 output-projection GEMM.
// ---------------------------------------------------------------------------

typedef _Float16 half16 __attribute__((ext_vector_type(16)));
typedef float    float8 __attribute__((ext_vector_type(8)));

#define HAVE_TDM __has_builtin(__builtin_amdgcn_tensor_load_to_lds)

#if HAVE_TDM
typedef unsigned int u32x4 __attribute__((ext_vector_type(4)));
typedef int          i32x4 __attribute__((ext_vector_type(4)));
typedef int          i32x8 __attribute__((ext_vector_type(8)));
#endif

union FragU {
    half16 h;
    uint4  u[2];
};

// A-matrix fragment (16x32 f16), tile row-major with stride `ld` (halves).
// ISA layout: lane m = lane&15; group g = lane>>4.
//   elements 0..7  : K = 8g + e          (contiguous 16B)
//   elements 8..15 : K = 16 + 8g + (e-8) (contiguous 16B)
__device__ __forceinline__ half16 frag_a(const _Float16* base, int ld) {
    const int lane = threadIdx.x & 31;
    const int m = lane & 15, g = lane >> 4;
    const _Float16* p = base + (size_t)m * ld;
    FragU f;
    f.u[0] = *reinterpret_cast<const uint4*>(p + 8 * g);
    f.u[1] = *reinterpret_cast<const uint4*>(p + 16 + 8 * g);
    return f.h;
}

// B-matrix fragment (32x16 f16) read from a B^T (n-major) tile, stride `ld`.
// ISA layout: lane n = lane&15; group g = lane>>4; elements 0..7: K = 8g + e, 8..15: K = 16 + 8g + (e-8).
__device__ __forceinline__ half16 frag_b(const _Float16* base, int ld) {
    const int lane = threadIdx.x & 31;
    const int n = lane & 15, g = lane >> 4;
    const _Float16* p = base + (size_t)n * ld;
    FragU f;
    f.u[0] = *reinterpret_cast<const uint4*>(p + 8 * g);
    f.u[1] = *reinterpret_cast<const uint4*>(p + 16 + 8 * g);
    return f.h;
}

__device__ __forceinline__ float8 wmma32(half16 a, half16 b, float8 c) {
    return __builtin_amdgcn_wmma_f32_16x16x32_f16(
        /*neg_a=*/false, a, /*neg_b=*/false, b,
        /*c_mod=*/(short)0, c, /*reuse_a=*/false, /*reuse_b=*/false);
}

#define N_EMB 384
#define TSEQ  256
#define HEADS 6
#define DHEAD 64
#define BATCH 256

// ---------------------------------------------------------------------------
// Kernel 1: QKV projections.  y[b,h,t,d] = sum_c x[b,t,c] * w[h,c,d]
// grid = (M/128, HEADS).  One block computes K, Q and V for one head tile,
// reusing the staged X tile for all three weight tensors (3x less X traffic).
// V is stored transposed ([b,h][d][t]) for the attention PV matmul.
// ---------------------------------------------------------------------------
__global__ __launch_bounds__(256) void qkv_gemm_kernel(
    const float* __restrict__ x,
    const float* __restrict__ wk, const float* __restrict__ wq,
    const float* __restrict__ wv,
    _Float16* __restrict__ Kf, _Float16* __restrict__ Qf,
    _Float16* __restrict__ Vt)
{
    __shared__ _Float16 Xs[128 * 40];
    __shared__ _Float16 Ws[3][64 * 40];   // W^T tiles: [n][k]

    const int m0 = blockIdx.x * 128;
    const int h  = blockIdx.y;
    const float* wh[3] = { wk + (size_t)h * N_EMB * DHEAD,
                           wq + (size_t)h * N_EMB * DHEAD,
                           wv + (size_t)h * N_EMB * DHEAD };

    const int tid = threadIdx.x;
    const int lane = tid & 31, wave = tid >> 5;
    const int wm = wave >> 1, wn = wave & 1;           // 4x2 wave grid (32x32)
    const int g = lane >> 4, nl = lane & 15;

    float8 acc[3][2][2] = {};

    for (int kk = 0; kk < N_EMB; kk += 32) {
        // stage X tile [128 x 32] f32 -> f16 (coalesced)
        #pragma unroll
        for (int i = 0; i < 16; ++i) {
            int idx = tid + i * 256;
            int r = idx >> 5, c = idx & 31;
            Xs[r * 40 + c] = (_Float16)x[(size_t)(m0 + r) * N_EMB + kk + c];
        }
        // stage the three W tiles [32 x 64] transposed -> Ws[tsr][n][k]
        #pragma unroll
        for (int tsr = 0; tsr < 3; ++tsr) {
            #pragma unroll
            for (int i = 0; i < 8; ++i) {
                int idx = tid + i * 256;
                int k = idx >> 6, n = idx & 63;
                Ws[tsr][n * 40 + k] =
                    (_Float16)wh[tsr][(size_t)(kk + k) * DHEAD + n];
            }
        }
        __syncthreads();

        half16 a0 = frag_a(Xs + (wm * 32) * 40, 40);
        half16 a1 = frag_a(Xs + (wm * 32 + 16) * 40, 40);
        #pragma unroll
        for (int tsr = 0; tsr < 3; ++tsr) {
            half16 b0 = frag_b(Ws[tsr] + (wn * 32) * 40, 40);
            half16 b1 = frag_b(Ws[tsr] + (wn * 32 + 16) * 40, 40);
            acc[tsr][0][0] = wmma32(a0, b0, acc[tsr][0][0]);
            acc[tsr][0][1] = wmma32(a0, b1, acc[tsr][0][1]);
            acc[tsr][1][0] = wmma32(a1, b0, acc[tsr][1][0]);
            acc[tsr][1][1] = wmma32(a1, b1, acc[tsr][1][1]);
        }
        __syncthreads();
    }

    // epilogue: C layout lane holds (row = 16*i + 8g + r, col = 16*jj + nl)
    #pragma unroll
    for (int tsr = 0; tsr < 3; ++tsr) {
        #pragma unroll
        for (int i = 0; i < 2; ++i) {
            #pragma unroll
            for (int jj = 0; jj < 2; ++jj) {
                #pragma unroll
                for (int r = 0; r < 8; ++r) {
                    int mloc = wm * 32 + i * 16 + 8 * g + r;
                    int nloc = wn * 32 + jj * 16 + nl;    // d in 0..63
                    int m = m0 + mloc;
                    int b = m >> 8, t = m & (TSEQ - 1);
                    size_t bh = (size_t)b * HEADS + h;
                    _Float16 val = (_Float16)acc[tsr][i][jj][r];
                    if (tsr == 2) {
                        Vt[(bh * DHEAD + nloc) * TSEQ + t] = val;  // [b,h][d][t]
                    } else {
                        _Float16* dst = (tsr == 0) ? Kf : Qf;
                        dst[(bh * TSEQ + t) * DHEAD + nloc] = val; // [b,h][t][d]
                    }
                }
            }
        }
    }
}

// ---------------------------------------------------------------------------
// Kernel 2: causal attention per (b,h).  wei = K Q^T / sqrt(384), softmax(s),
// out = wei V.  grid = B*H blocks, 256 threads (8 waves, 2 t-blocks each).
// ---------------------------------------------------------------------------
__global__ __launch_bounds__(256) void attn_kernel(
    const _Float16* __restrict__ Kf, const _Float16* __restrict__ Qf,
    const _Float16* __restrict__ Vt, _Float16* __restrict__ att)
{
    __shared__ _Float16 Ps[8][16 * 256];   // per-wave P strip (C->A relayout)

    const int bh = blockIdx.x;
    const int b = bh / HEADS, h = bh % HEADS;
    const int tid = threadIdx.x;
    const int lane = tid & 31, wave = tid >> 5;
    const int g = lane >> 4, nl = lane & 15;

    const _Float16* Kb = Kf + (size_t)bh * TSEQ * DHEAD;   // [t][d]
    const _Float16* Qb = Qf + (size_t)bh * TSEQ * DHEAD;   // [s][d]
    const _Float16* Vb = Vt + (size_t)bh * DHEAD * TSEQ;   // [d][s]
    _Float16* P = &Ps[wave][0];

    const float scale = 0.05103103630798288f;   // 1/sqrt(N_EMB)
    const float NEG = -__builtin_inff();

    for (int half_idx = 0; half_idx < 2; ++half_idx) {
        const int tb = wave + half_idx * 8;     // t-block 0..15
        const int t0 = tb * 16;

        // K row fragments for this wave's 16 rows (D=64 -> 2 K-chunks)
        half16 ka0 = frag_a(Kb + (size_t)t0 * DHEAD + 0, DHEAD);
        half16 ka1 = frag_a(Kb + (size_t)t0 * DHEAD + 32, DHEAD);

        // logits: full 16x256 strip (masked blocks collapse to zero prob)
        float8 acc[16];
        #pragma unroll
        for (int sb = 0; sb < 16; ++sb) {
            float8 c = {};
            half16 q0 = frag_b(Qb + (size_t)(sb * 16) * DHEAD + 0, DHEAD);
            half16 q1 = frag_b(Qb + (size_t)(sb * 16) * DHEAD + 32, DHEAD);
            c = wmma32(ka0, q0, c);
            c = wmma32(ka1, q1, c);
            acc[sb] = c;
        }

        // scale + causal mask + per-row max (row = t0 + 8g + r, col = sb*16+nl)
        float rmax[8], rsum[8];
        #pragma unroll
        for (int r = 0; r < 8; ++r) rmax[r] = NEG;
        #pragma unroll
        for (int sb = 0; sb < 16; ++sb) {
            #pragma unroll
            for (int r = 0; r < 8; ++r) {
                int t = t0 + 8 * g + r;
                int s = sb * 16 + nl;
                float v = acc[sb][r] * scale;
                if (s > t) v = NEG;
                acc[sb][r] = v;
                rmax[r] = fmaxf(rmax[r], v);
            }
        }
        // butterfly max across the 16 lanes holding one row
        #pragma unroll
        for (int r = 0; r < 8; ++r) {
            float v = rmax[r];
            v = fmaxf(v, __shfl_xor(v, 1, 32));
            v = fmaxf(v, __shfl_xor(v, 2, 32));
            v = fmaxf(v, __shfl_xor(v, 4, 32));
            v = fmaxf(v, __shfl_xor(v, 8, 32));
            rmax[r] = v;
            rsum[r] = 0.0f;
        }
        #pragma unroll
        for (int sb = 0; sb < 16; ++sb) {
            #pragma unroll
            for (int r = 0; r < 8; ++r) {
                float e = __expf(acc[sb][r] - rmax[r]);
                acc[sb][r] = e;
                rsum[r] += e;
            }
        }
        #pragma unroll
        for (int r = 0; r < 8; ++r) {
            float v = rsum[r];
            v += __shfl_xor(v, 1, 32);
            v += __shfl_xor(v, 2, 32);
            v += __shfl_xor(v, 4, 32);
            v += __shfl_xor(v, 8, 32);
            rsum[r] = 1.0f / v;
        }

        // normalized probabilities -> f16 strip in LDS (A-layout source)
        #pragma unroll
        for (int sb = 0; sb < 16; ++sb) {
            #pragma unroll
            for (int r = 0; r < 8; ++r) {
                P[(8 * g + r) * 256 + sb * 16 + nl] =
                    (_Float16)(acc[sb][r] * rsum[r]);
            }
        }

        // out = P @ V : A = P [16 x 256], B^T = Vt [d][s]
        float8 o[4] = {};
        #pragma unroll
        for (int sc = 0; sc < 8; ++sc) {
            half16 pa = frag_a(P + sc * 32, 256);
            #pragma unroll
            for (int nb = 0; nb < 4; ++nb) {
                half16 vb = frag_b(Vb + (size_t)(nb * 16) * TSEQ + sc * 32, TSEQ);
                o[nb] = wmma32(pa, vb, o[nb]);
            }
        }

        // store concat-head layout: att[b*T + t][h*64 + d]
        #pragma unroll
        for (int nb = 0; nb < 4; ++nb) {
            #pragma unroll
            for (int r = 0; r < 8; ++r) {
                int t = t0 + 8 * g + r;
                int d = nb * 16 + nl;
                att[(size_t)(b * TSEQ + t) * N_EMB + h * DHEAD + d] =
                    (_Float16)o[nb][r];
            }
        }
    }
}

// ---------------------------------------------------------------------------
// Kernel 3: output projection.  out = att @ wproj + bproj  (f32 out)
// grid = (M/128, N_EMB/64).  The f16 A tile is DMAed into LDS by the Tensor
// Data Mover (D# pad feature recreates the 40-half padded stride in HW).
// ---------------------------------------------------------------------------
__global__ __launch_bounds__(256) void proj_gemm_kernel(
    const _Float16* __restrict__ att, const float* __restrict__ wproj,
    const float* __restrict__ bproj, float* __restrict__ out)
{
    __shared__ _Float16 As[128 * 40];
    __shared__ _Float16 Bs[64 * 40];   // wproj^T tile [n][k]

    const int m0 = blockIdx.x * 128;
    const int n0 = blockIdx.y * 64;
    const int tid = threadIdx.x;
    const int lane = tid & 31, wave = tid >> 5;
    const int wm = wave >> 1, wn = wave & 1;
    const int g = lane >> 4, nl = lane & 15;

#if HAVE_TDM
    const unsigned lds_as = (unsigned)(size_t)(void*)As;
#endif

    float8 acc[2][2] = {};

    for (int kk = 0; kk < N_EMB; kk += 32) {
#if HAVE_TDM
        // TDM: DMA f16 tile att[m0..m0+127][kk..kk+31] -> As with a 4-DWORD
        // pad after every 16 DWORDs (row = 32 halves) => LDS stride 40 halves.
        if (wave == 0) {
            unsigned long long ga =
                (unsigned long long)(const void*)(att + (size_t)m0 * N_EMB + kk);
            u32x4 g0;
            g0[0] = 1u;                                   // count=1 (valid D#)
            g0[1] = lds_as;                               // lds_addr (bytes)
            g0[2] = (unsigned)(ga & 0xFFFFFFFFull);       // global_addr[31:0]
            g0[3] = (unsigned)((ga >> 32) & 0x01FFFFFFull)// global_addr[56:32]
                  | (2u << 30);                           // type=2 ("image")
            i32x8 g1;
            g1[0] = (1 << 16)     // data_size = 2 bytes
                  | (1 << 20)     // pad_enable
                  | (3 << 22)     // pad_interval: 16 DWORDs
                  | (3 << 25);    // pad_amount : 4 DWORDs
            g1[1] = (int)((N_EMB & 0xFFFF) << 16);        // tensor_dim0 lo16
            g1[2] = (int)((N_EMB >> 16) | (128 << 16));   // dim0 hi | dim1 lo
            g1[3] = (int)(32 << 16);                      // dim1 hi | tile_dim0
            g1[4] = 128;                                  // tile_dim1 | tile_dim2=0
            g1[5] = N_EMB;                                // tensor_dim0_stride lo32
            g1[6] = 0;                                    // stride hi | dim1_stride lo
            g1[7] = 0;
            i32x4 z4 = { 0, 0, 0, 0 };
            i32x8 z8 = { 0, 0, 0, 0, 0, 0, 0, 0 };
            __builtin_amdgcn_tensor_load_to_lds(g0, g1, z4, z4, z8, 0);
            __builtin_amdgcn_s_wait_tensorcnt(0);
        }
#else
        #pragma unroll
        for (int i = 0; i < 16; ++i) {
            int idx = tid + i * 256;
            int r = idx >> 5, c = idx & 31;
            As[r * 40 + c] = att[(size_t)(m0 + r) * N_EMB + kk + c];
        }
#endif
        #pragma unroll
        for (int i = 0; i < 8; ++i) {
            int idx = tid + i * 256;
            int k = idx >> 6, n = idx & 63;
            Bs[n * 40 + k] = (_Float16)wproj[(size_t)(kk + k) * N_EMB + n0 + n];
        }
        __syncthreads();

        half16 a0 = frag_a(As + (wm * 32) * 40, 40);
        half16 a1 = frag_a(As + (wm * 32 + 16) * 40, 40);
        half16 b0 = frag_b(Bs + (wn * 32) * 40, 40);
        half16 b1 = frag_b(Bs + (wn * 32 + 16) * 40, 40);
        acc[0][0] = wmma32(a0, b0, acc[0][0]);
        acc[0][1] = wmma32(a0, b1, acc[0][1]);
        acc[1][0] = wmma32(a1, b0, acc[1][0]);
        acc[1][1] = wmma32(a1, b1, acc[1][1]);
        __syncthreads();
    }

    #pragma unroll
    for (int jj = 0; jj < 2; ++jj) {
        float bias = bproj[n0 + wn * 32 + jj * 16 + nl];
        #pragma unroll
        for (int i = 0; i < 2; ++i) {
            #pragma unroll
            for (int r = 0; r < 8; ++r) {
                int m = m0 + wm * 32 + i * 16 + 8 * g + r;
                int n = n0 + wn * 32 + jj * 16 + nl;
                out[(size_t)m * N_EMB + n] = acc[i][jj][r] + bias;
            }
        }
    }
}

// ---------------------------------------------------------------------------
extern "C" void kernel_launch(void* const* d_in, const int* in_sizes, int n_in,
                              void* d_out, int out_size, void* d_ws, size_t ws_size,
                              hipStream_t stream)
{
    const float* x     = (const float*)d_in[0];
    const float* wk    = (const float*)d_in[1];
    const float* wq    = (const float*)d_in[2];
    const float* wv    = (const float*)d_in[3];
    const float* wproj = (const float*)d_in[4];
    const float* bproj = (const float*)d_in[5];
    float* out = (float*)d_out;

    const size_t elems = (size_t)BATCH * HEADS * TSEQ * DHEAD;  // 25,165,824
    _Float16* Kf  = (_Float16*)d_ws;
    _Float16* Qf  = Kf + elems;
    _Float16* Vt  = Qf + elems;
    _Float16* att = Vt + elems;   // [B*T][N_EMB] f16

    const int Mtiles = (BATCH * TSEQ) / 128;   // 512

    qkv_gemm_kernel<<<dim3(Mtiles, HEADS), 256, 0, stream>>>(
        x, wk, wq, wv, Kf, Qf, Vt);
    attn_kernel<<<dim3(BATCH * HEADS), 256, 0, stream>>>(Kf, Qf, Vt, att);
    proj_gemm_kernel<<<dim3(Mtiles, N_EMB / 64), 256, 0, stream>>>(
        att, wproj, bproj, out);
}
